// ExampleBinaryNet_41231686042119
// MI455X (gfx1250) — matrix operand order repaired and checked by reading, not compiled, not run on hardware
//
#include <hip/hip_runtime.h>

// Binarized LeNet forward on gfx1250 via f16 WMMA (f32 accumulate).
// All GEMM A-operands are laid out so each lane's fragment is 2-4 contiguous
// vector loads (halves 0-7 of a lane = K run of 8; halves 8-15 = run at +16),
// eliminating per-element im2col gather address math.

typedef __attribute__((ext_vector_type(16))) _Float16 v16h;
typedef __attribute__((ext_vector_type(8)))  _Float16 v8h;
typedef __attribute__((ext_vector_type(4)))  _Float16 v4h;
typedef __attribute__((ext_vector_type(8)))  float    v8f;

union AFrag {
    v16h v;
    v8h  o[2];
    v4h  q[4];
};

// ---- Weight pack: emit sign(w) as f16 WMMA B fragments.
//      Fragment (nt,kt) at offset (nt*Ktiles+kt)*512; element lane*16+h with
//      N = nt*16 + lane%16, logical K = kt*32 + 8*(lane/16) + (h&7) + 16*(h>>3).
//      mode 0: plain [N][Ksrc] row-major
//      mode 1: conv1 reorder  k = (ky*8+kx)*4 + ci   (ci<3, kx<5 valid)
//      mode 2: conv2 reorder  k = (ky*5+kx)*128 + ci (ci<100 valid)
__global__ void bnn_pack_b(const float* __restrict__ w, _Float16* __restrict__ out,
                           int N, int Ksrc, int Ntiles, int Ktiles, int mode) {
    int idx = blockIdx.x * blockDim.x + threadIdx.x;
    int total = Ntiles * Ktiles * 512;
    if (idx >= total) return;
    int frag = idx >> 9;
    int e    = idx & 511;
    int lane = e >> 4;
    int h    = e & 15;
    int kt = frag % Ktiles;
    int nt = frag / Ktiles;
    int n = nt * 16 + (lane & 15);
    int k = kt * 32 + ((lane >> 4) << 3) + (h & 7) + ((h >> 3) << 4);

    float wv = 0.f;
    bool ok = false;
    if (n < N) {
        if (mode == 0) {
            if (k < Ksrc) { wv = w[(size_t)n * Ksrc + k]; ok = true; }
        } else if (mode == 1) {
            int ci = k & 3, pos = k >> 2, ky = pos >> 3, kx = pos & 7;
            if (ci < 3 && kx < 5) { wv = w[n * 75 + ci * 25 + ky * 5 + kx]; ok = true; }
        } else {
            int ci = k & 127, pos = k >> 7;
            if (ci < 100) { wv = w[n * 2500 + ci * 25 + pos]; ok = true; }
        }
    }
    float val = ok ? ((wv > 0.f) ? 1.f : ((wv < 0.f) ? -1.f : 0.f)) : 0.f;
    out[idx] = (_Float16)val;
}

// ---- x transpose: x[2048,3,32,32] f32 -> xt[b][iy 32][ix 40 pad][ci 4 pad] f16
__global__ void bnn_xt(const float* __restrict__ x, _Float16* __restrict__ xt, int total) {
    int idx = blockIdx.x * blockDim.x + threadIdx.x;
    if (idx >= total) return;
    int ci = idx & 3;
    int t  = idx >> 2;
    int ix = t % 40; t /= 40;
    int iy = t & 31;
    int b  = t >> 5;
    float v = 0.f;
    if (ci < 3 && ix < 32) v = x[(((size_t)b * 3 + ci) * 32 + iy) * 32 + ix];
    xt[idx] = (_Float16)v;
}

__global__ void bnn_zero8(_Float16* __restrict__ p, int n8) {
    int i = blockIdx.x * blockDim.x + threadIdx.x;
    if (i < n8) { v8h z = {}; ((v8h*)p)[i] = z; }
}

// ---- conv1: implicit GEMM, A from xt (K=(ky*8+kx)*4+ci, Kpad=160, 5 ksteps),
//      7 N-tiles (100->112). Fused bias + hardtanh + within-lane 2x2 maxpool.
//      Output h1c[b][py 14][px 14][ci 128 pad] f16 (padding pre-zeroed).
__global__ __launch_bounds__(256) void bnn_conv1_wmma(
    const _Float16* __restrict__ xt, const _Float16* __restrict__ Bp,
    const float* __restrict__ bias, _Float16* __restrict__ h1c, int Mtiles) {
    int gw = blockIdx.x * (blockDim.x >> 5) + (threadIdx.x >> 5);
    if (gw >= Mtiles) return;
    int lane = threadIdx.x & 31;
    int r = lane & 15;                 // A/C row within tile
    int pp  = gw * 4 + (r >> 2);       // pooled pixel index
    int sub = r & 3;                   // 2x2 sub-position
    int bidx = pp / 196;
    int rem  = pp - bidx * 196;
    int py = rem / 14, px = rem - py * 14;
    int oy = 2 * py + (sub >> 1);
    int ox = 2 * px + (sub & 1);
    int kx0 = (lane >> 4) << 1;        // 0 or 2
    const _Float16* xtb = xt + (size_t)bidx * 32 * 40 * 4;

    v8f zero = {};
    v8f acc[7];
#pragma unroll
    for (int nt = 0; nt < 7; ++nt) acc[nt] = zero;

#pragma unroll
    for (int kk = 0; kk < 5; ++kk) {   // kk == ky
        AFrag af;
        const _Float16* prow = xtb + (((size_t)(oy + kk) * 40 + ox + kx0) << 2);
        af.q[0] = *(const v4h*)(prow);       // kx0,   ci0..3 | kx0+1 (first 4)
        af.q[1] = *(const v4h*)(prow + 4);
        af.q[2] = *(const v4h*)(prow + 16);  // kx0+4, kx0+5
        af.q[3] = *(const v4h*)(prow + 20);
#pragma unroll
        for (int nt = 0; nt < 7; ++nt) {
            v16h bf = ((const v16h*)(Bp + (((size_t)nt * 5 + kk) << 9)))[lane];
            acc[nt] = __builtin_amdgcn_wmma_f32_16x16x32_f16(
                false, af.v, false, bf, (short)0, acc[nt], false, false);
        }
    }

    int n0 = lane & 15;
    int gbase = (lane >> 4) << 1;
#pragma unroll
    for (int nt = 0; nt < 7; ++nt) {
        int n = nt * 16 + n0;
        if (n >= 100) continue;        // h1c padding stays zero (pre-zeroed)
        float bv = bias[n];
        float m0 = -2.f, m1 = -2.f;
#pragma unroll
        for (int i = 0; i < 8; ++i) {
            float y = acc[nt][i] + bv;
            y = fminf(1.f, fmaxf(-1.f, y));
            if (i < 4) m0 = fmaxf(m0, y); else m1 = fmaxf(m1, y);
        }
        float mx[2] = {m0, m1};
#pragma unroll
        for (int j = 0; j < 2; ++j) {
            int ppg = gw * 4 + gbase + j;
            int bb = ppg / 196;
            int r2 = ppg - bb * 196;
            h1c[((size_t)(bb * 196 + r2) << 7) + n] = (_Float16)mx[j];
        }
    }
}

// ---- conv2: A from h1c (K = pos*128+ci, Kpad=3200, 100 ksteps), N=16.
//      Fused bias + hardtanh + maxpool -> h2[b][416] (ci*25+py*5+px, padded).
__global__ __launch_bounds__(256) void bnn_conv2_wmma(
    const _Float16* __restrict__ h1c, const _Float16* __restrict__ Bp,
    const float* __restrict__ bias, _Float16* __restrict__ h2, int Mtiles) {
    int gw = blockIdx.x * (blockDim.x >> 5) + (threadIdx.x >> 5);
    if (gw >= Mtiles) return;
    int lane = threadIdx.x & 31;
    int r = lane & 15;
    int pp  = gw * 4 + (r >> 2);
    int sub = r & 3;
    int bidx = pp / 25;
    int rem  = pp - bidx * 25;
    int py = rem / 5, px = rem - py * 5;
    int oy = 2 * py + (sub >> 1);
    int ox = 2 * px + (sub & 1);
    int kgrp = (lane >> 4) << 3;       // 0 or 8
    const _Float16* hb = h1c + ((size_t)bidx * 196 << 7);

    v8f acc = {};
#pragma unroll 5
    for (int pos = 0; pos < 25; ++pos) {
        int ky = pos / 5, kx = pos - (pos / 5) * 5;
        const _Float16* hp = hb + ((size_t)((oy + ky) * 14 + (ox + kx)) << 7) + kgrp;
#pragma unroll
        for (int j = 0; j < 4; ++j) {
            AFrag af;
            af.o[0] = *(const v8h*)(hp + j * 32);
            af.o[1] = *(const v8h*)(hp + j * 32 + 16);
            v16h bf = ((const v16h*)(Bp + ((size_t)(pos * 4 + j) << 9)))[lane];
            acc = __builtin_amdgcn_wmma_f32_16x16x32_f16(
                false, af.v, false, bf, (short)0, acc, false, false);
        }
    }

    int n = lane & 15;
    float bv = bias[n];
    int gbase = (lane >> 4) << 1;
    float m0 = -2.f, m1 = -2.f;
#pragma unroll
    for (int i = 0; i < 8; ++i) {
        float y = acc[i] + bv;
        y = fminf(1.f, fmaxf(-1.f, y));
        if (i < 4) m0 = fmaxf(m0, y); else m1 = fmaxf(m1, y);
    }
    float mx[2] = {m0, m1};
#pragma unroll
    for (int j = 0; j < 2; ++j) {
        int ppg = gw * 4 + gbase + j;
        int bb = ppg / 25;
        int r2 = ppg - bb * 25;
        int pyy = r2 / 5, pxx = r2 - (r2 / 5) * 5;
        h2[(size_t)bb * 416 + n * 25 + pyy * 5 + pxx] = (_Float16)mx[j];
    }
}

// ---- Generic FC GEMM: out[m][n] = act(A[m][:] . signW[n][:] + bias[n]).
//      A f16 row-major [2048][lda], K-padded with zeros. One wave per
//      (Mtile, Ntile); grid.x = 128 M-tiles, wave id = N-tile.
__global__ __launch_bounds__(256) void bnn_fc_wmma(
    const _Float16* __restrict__ A, int lda,
    const _Float16* __restrict__ Bp, const float* __restrict__ bias,
    int Ktiles, int Ntiles, int Nreal, int act,
    _Float16* __restrict__ outH, float* __restrict__ outF, int ldo) {
    int wave = threadIdx.x >> 5;
    if (wave >= Ntiles) return;
    int lane = threadIdx.x & 31;
    int nt = wave;
    int mt = blockIdx.x;
    int m = mt * 16 + (lane & 15);
    int kgrp = (lane >> 4) << 3;
    const _Float16* Arow = A + (size_t)m * lda + kgrp;

    v8f acc = {};
    for (int kk = 0; kk < Ktiles; ++kk) {
        AFrag af;
        af.o[0] = *(const v8h*)(Arow + kk * 32);
        af.o[1] = *(const v8h*)(Arow + kk * 32 + 16);
        v16h bf = ((const v16h*)(Bp + (((size_t)nt * Ktiles + kk) << 9)))[lane];
        acc = __builtin_amdgcn_wmma_f32_16x16x32_f16(
            false, af.v, false, bf, (short)0, acc, false, false);
    }

    int n = nt * 16 + (lane & 15);
    float bv = (n < Nreal) ? bias[n] : 0.f;
    int base = (lane >> 4) << 3;       // C rows: i + 8*(lane/16)
#pragma unroll
    for (int i = 0; i < 8; ++i) {
        float y = acc[i] + bv;
        if (act) y = fminf(1.f, fmaxf(-1.f, y));
        int mrow = mt * 16 + base + i;
        if (outH) {
            outH[(size_t)mrow * ldo + n] = (n < Nreal) ? (_Float16)y : (_Float16)0.f;
        } else if (n < Nreal) {
            outF[(size_t)mrow * ldo + n] = y;
        }
    }
}

extern "C" void kernel_launch(void* const* d_in, const int* in_sizes, int n_in,
                              void* d_out, int out_size, void* d_ws, size_t ws_size,
                              hipStream_t stream) {
    (void)in_sizes; (void)n_in; (void)out_size; (void)ws_size;
    const float* x  = (const float*)d_in[0];
    const float* w1 = (const float*)d_in[1];
    const float* b1 = (const float*)d_in[2];
    const float* w2 = (const float*)d_in[3];
    const float* b2 = (const float*)d_in[4];
    const float* w3 = (const float*)d_in[5];
    const float* b3 = (const float*)d_in[6];
    const float* w4 = (const float*)d_in[7];
    const float* b4 = (const float*)d_in[8];
    const float* w5 = (const float*)d_in[9];
    const float* b5 = (const float*)d_in[10];

    // workspace carve-out (256B aligned)
    size_t off = 0;
    char* ws = (char*)d_ws;
    auto take = [&](size_t bytes) -> void* {
        void* p = ws + off;
        off += (bytes + 255) & ~(size_t)255;
        return p;
    };
    _Float16* xt  = (_Float16*)take((size_t)2048 * 32 * 40 * 4 * 2);   //  21 MB
    _Float16* h1c = (_Float16*)take((size_t)2048 * 196 * 128 * 2);     // 103 MB
    _Float16* h2  = (_Float16*)take((size_t)2048 * 416 * 2);
    _Float16* h3  = (_Float16*)take((size_t)2048 * 128 * 2);
    _Float16* h4  = (_Float16*)take((size_t)2048 * 96  * 2);
    _Float16* w1p = (_Float16*)take((size_t)7   * 5   * 512 * 2);
    _Float16* w2p = (_Float16*)take((size_t)1   * 100 * 512 * 2);
    _Float16* w3p = (_Float16*)take((size_t)8   * 13  * 512 * 2);
    _Float16* w4p = (_Float16*)take((size_t)6   * 4   * 512 * 2);
    _Float16* w5p = (_Float16*)take((size_t)1   * 3   * 512 * 2);

    // pack sign-binarized weights into WMMA B fragments (per-layer K order)
    auto pk = [&](const float* w, _Float16* o, int N, int Ksrc, int Nt, int Kt, int mode) {
        int total = Nt * Kt * 512;
        bnn_pack_b<<<(total + 255) / 256, 256, 0, stream>>>(w, o, N, Ksrc, Nt, Kt, mode);
    };
    pk(w1, w1p, 100, 75,   7, 5,   1);
    pk(w2, w2p, 16,  2500, 1, 100, 2);
    pk(w3, w3p, 120, 400,  8, 13,  0);
    pk(w4, w4p, 84,  120,  6, 4,   0);
    pk(w5, w5p, 10,  84,   1, 3,   0);

    // x -> channel-last padded f16
    {
        int total = 2048 * 32 * 40 * 4;
        bnn_xt<<<(total + 255) / 256, 256, 0, stream>>>(x, xt, total);
    }
    // zero padded activation buffers (deterministic every call)
    {
        int n8 = 2048 * 196 * 128 / 8;
        bnn_zero8<<<(n8 + 255) / 256, 256, 0, stream>>>(h1c, n8);
        int m8 = 2048 * 416 / 8;
        bnn_zero8<<<(m8 + 255) / 256, 256, 0, stream>>>(h2, m8);
    }

    // conv1: 2048*14*14 pooled pixels * 4 subs / 16 rows = 100352 M-tiles
    bnn_conv1_wmma<<<100352 / 8, 256, 0, stream>>>(xt, w1p, b1, h1c, 100352);
    // conv2: 2048*5*5 pooled pixels * 4 / 16 = 12800 M-tiles
    bnn_conv2_wmma<<<12800 / 8, 256, 0, stream>>>(h1c, w2p, b2, h2, 12800);

    // FC stack: 2048 rows = 128 M-tiles
    bnn_fc_wmma<<<128, 256, 0, stream>>>(h2, 416, w3p, b3, 13, 8, 120, 1,
                                         h3, nullptr, 128);
    bnn_fc_wmma<<<128, 256, 0, stream>>>(h3, 128, w4p, b4, 4, 6, 84, 1,
                                         h4, nullptr, 96);
    bnn_fc_wmma<<<128, 256, 0, stream>>>(h4, 96, w5p, b5, 3, 1, 10, 0,
                                         nullptr, (float*)d_out, 10);
}
